// DeeperGCN_68693706932335
// MI455X (gfx1250) — matrix operand order repaired and checked by reading, not compiled, not run on hardware
//
#include <hip/hip_runtime.h>
#include <hip/hip_bf16.h>

#define NN 50000
#define EE 600000
#define CC 128
#define LL 7
#define NGG 512
#define EPSF 1e-7f

typedef _Float16 v16h __attribute__((ext_vector_type(16)));
typedef _Float16 v8h  __attribute__((ext_vector_type(8)));
typedef float    v8f  __attribute__((ext_vector_type(8)));

// ---------------- elementwise / fill ----------------
__global__ void k_fill(float* __restrict__ p, float v, int n) {
    int i = blockIdx.x * blockDim.x + threadIdx.x;
    if (i < n) p[i] = v;
}

__global__ void k_add_inplace(float* __restrict__ a, const float* __restrict__ b, int n) {
    int i = blockIdx.x * blockDim.x + threadIdx.x;
    if (i < n) a[i] += b[i];
}

__global__ void k_convert(const float* __restrict__ s, _Float16* __restrict__ d, int n) {
    int i = blockIdx.x * blockDim.x + threadIdx.x;
    if (i < n) d[i] = (_Float16)s[i];
}

// zf16 = f16(h + m)   (z = hx + aggregated message, staged for GEMM1)
__global__ void k_add_convert(const float* __restrict__ h, const float* __restrict__ m,
                              _Float16* __restrict__ z, int n) {
    int i = blockIdx.x * blockDim.x + threadIdx.x;
    if (i < n) z[i] = (_Float16)(h[i] + m[i]);
}

// h2[n,c] += vn[batch[n], c]
__global__ void k_add_gather(float* __restrict__ h2, const float* __restrict__ vn,
                             const int* __restrict__ batch, int n) {
    int i = blockIdx.x * blockDim.x + threadIdx.x;
    if (i < n) h2[i] += vn[(size_t)batch[i / CC] * CC + (i % CC)];
}

// ---------------- encoders ----------------
__global__ void k_atom_encode(const int* __restrict__ x, const float* __restrict__ emb,
                              const float* __restrict__ vn_emb, float* __restrict__ h) {
    int i = blockIdx.x * blockDim.x + threadIdx.x;
    if (i >= NN * CC) return;
    int n = i / CC, c = i % CC;
    float acc = vn_emb[c];                       // initial vn broadcast folded in
    #pragma unroll
    for (int f = 0; f < 9; ++f)
        acc += emb[((size_t)(f * 64 + x[n * 9 + f])) * CC + c];
    h[i] = acc;
}

__global__ void k_bond_encode(const int* __restrict__ ea, const float* __restrict__ emb,
                              float* __restrict__ ee) {
    int i = blockIdx.x * blockDim.x + threadIdx.x;
    if (i >= EE * CC) return;
    int e = i / CC, c = i % CC;
    float acc = 0.f;
    #pragma unroll
    for (int f = 0; f < 3; ++f)
        acc += emb[((size_t)(f * 8 + ea[e * 3 + f])) * CC + c];
    ee[(size_t)e * CC + c] = acc;
}

__global__ void k_bcast_vn(const float* __restrict__ vn_emb, float* __restrict__ vn) {
    int i = blockIdx.x * blockDim.x + threadIdx.x;
    if (i < NGG * CC) vn[i] = vn_emb[i % CC];
}

// ---------------- message passing (msg recomputed; node arrays are L2-resident) ----------------
__global__ void k_msg_max(const float* __restrict__ hx, const float* __restrict__ ee,
                          const int* __restrict__ src, const int* __restrict__ dst,
                          int* __restrict__ mx, int total) {
    int i = blockIdx.x * blockDim.x + threadIdx.x;
    if (i >= total) return;
    int e = i >> 5;               // C/4 = 32 quads per edge
    int c4 = (i & 31) << 2;
    int s = src[e], d = dst[e];
    float4 hv = *(const float4*)(hx + (size_t)s * CC + c4);
    float4 ev = *(const float4*)(ee + (size_t)e * CC + c4);
    float m0 = fmaxf(hv.x + ev.x, 0.f) + EPSF;
    float m1 = fmaxf(hv.y + ev.y, 0.f) + EPSF;
    float m2 = fmaxf(hv.z + ev.z, 0.f) + EPSF;
    float m3 = fmaxf(hv.w + ev.w, 0.f) + EPSF;
    int* p = mx + (size_t)d * CC + c4;
    atomicMax(p + 0, __float_as_int(m0));   // msg > 0 -> int compare is order-preserving
    atomicMax(p + 1, __float_as_int(m1));
    atomicMax(p + 2, __float_as_int(m2));
    atomicMax(p + 3, __float_as_int(m3));
}

__global__ void k_den(const float* __restrict__ hx, const float* __restrict__ ee,
                      const int* __restrict__ src, const int* __restrict__ dst,
                      const float* __restrict__ mx, float* __restrict__ den, int total) {
    int i = blockIdx.x * blockDim.x + threadIdx.x;
    if (i >= total) return;
    int e = i >> 5;
    int c4 = (i & 31) << 2;
    int s = src[e], d = dst[e];
    float4 hv = *(const float4*)(hx + (size_t)s * CC + c4);
    float4 ev = *(const float4*)(ee + (size_t)e * CC + c4);
    float4 xv = *(const float4*)(mx + (size_t)d * CC + c4);
    float* p = den + (size_t)d * CC + c4;
    atomicAdd(p + 0, __expf(fmaxf(hv.x + ev.x, 0.f) + EPSF - xv.x));
    atomicAdd(p + 1, __expf(fmaxf(hv.y + ev.y, 0.f) + EPSF - xv.y));
    atomicAdd(p + 2, __expf(fmaxf(hv.z + ev.z, 0.f) + EPSF - xv.z));
    atomicAdd(p + 3, __expf(fmaxf(hv.w + ev.w, 0.f) + EPSF - xv.w));
}

__global__ void k_msum(const float* __restrict__ hx, const float* __restrict__ ee,
                       const int* __restrict__ src, const int* __restrict__ dst,
                       const float* __restrict__ mx, const float* __restrict__ den,
                       float* __restrict__ m, int total) {
    int i = blockIdx.x * blockDim.x + threadIdx.x;
    if (i >= total) return;
    int e = i >> 5;
    int c4 = (i & 31) << 2;
    int s = src[e], d = dst[e];
    float4 hv = *(const float4*)(hx + (size_t)s * CC + c4);
    float4 ev = *(const float4*)(ee + (size_t)e * CC + c4);
    float4 xv = *(const float4*)(mx + (size_t)d * CC + c4);
    float4 dv = *(const float4*)(den + (size_t)d * CC + c4);
    float* p = m + (size_t)d * CC + c4;
    float g0 = fmaxf(hv.x + ev.x, 0.f) + EPSF;
    float g1 = fmaxf(hv.y + ev.y, 0.f) + EPSF;
    float g2 = fmaxf(hv.z + ev.z, 0.f) + EPSF;
    float g3 = fmaxf(hv.w + ev.w, 0.f) + EPSF;
    atomicAdd(p + 0, g0 * (__expf(g0 - xv.x) / (dv.x + 1e-16f)));
    atomicAdd(p + 1, g1 * (__expf(g1 - xv.y) / (dv.y + 1e-16f)));
    atomicAdd(p + 2, g2 * (__expf(g2 - xv.z) / (dv.z + 1e-16f)));
    atomicAdd(p + 3, g3 * (__expf(g3 - xv.w) / (dv.w + 1e-16f)));
}

// ---------------- segment sum over batch ----------------
__global__ void k_segsum(const float* __restrict__ h, const int* __restrict__ batch,
                         float* __restrict__ out, int n) {
    int i = blockIdx.x * blockDim.x + threadIdx.x;
    if (i < n) atomicAdd(&out[(size_t)batch[i / CC] * CC + (i % CC)], h[i]);
}

// ---------------- weight convert + transpose: W[K][Ncol] f32 -> Wt[Ncol][K] f16 ----------------
__global__ void k_wt(const float* __restrict__ W, _Float16* __restrict__ Wt, int K, int Ncol) {
    int i = blockIdx.x * blockDim.x + threadIdx.x;
    if (i >= K * Ncol) return;
    int n = i / K, k = i % K;
    Wt[i] = (_Float16)W[(size_t)k * Ncol + n];
}

// ---------------- LayerNorm (+opt ReLU), wave per row; writes f32 and/or f16 ----------------
__global__ void k_ln(const float* __restrict__ Z, const float* __restrict__ g,
                     const float* __restrict__ bb, float* __restrict__ outf,
                     _Float16* __restrict__ outh, int R, int D, int doRelu) {
    int wid = threadIdx.x >> 5;
    int lane = threadIdx.x & 31;
    int row = blockIdx.x * (blockDim.x >> 5) + wid;
    if (row >= R) return;
    const float* zr = Z + (size_t)row * D;
    int npl = D >> 5;                 // 4 (D=128) or 8 (D=256)
    float v[8];
    float s = 0.f;
    for (int i = 0; i < npl; ++i) { v[i] = zr[lane + (i << 5)]; s += v[i]; }
    for (int o = 16; o > 0; o >>= 1) s += __shfl_xor(s, o, 32);
    float mu = s / (float)D;
    float var = 0.f;
    for (int i = 0; i < npl; ++i) { float dd = v[i] - mu; var += dd * dd; }
    for (int o = 16; o > 0; o >>= 1) var += __shfl_xor(var, o, 32);
    float rs = rsqrtf(var / (float)D + 1e-5f);
    for (int i = 0; i < npl; ++i) {
        int c = lane + (i << 5);
        float y = (v[i] - mu) * rs * g[c] + bb[c];
        if (doRelu) y = fmaxf(y, 0.f);
        if (outf) outf[(size_t)row * D + c] = y;
        if (outh) outh[(size_t)row * D + c] = (_Float16)y;
    }
}

// ---------------- WMMA GEMM: D[M,Ncol] = A16[M,K] * Wt16[Ncol,K]^T + bias ----------------
// Block = 8 waves x 16 rows = 128 rows; blockIdx.y selects a 64-column slab.
// The 64xK f16 weight slab (contiguous in the transposed layout, <=32KB) is staged
// into LDS once per block via gfx1250 async global->LDS copies (ASYNCcnt), then each
// wave computes a 16x64 tile: per k-step one A fragment (2x b128) feeds 4 WMMAs whose
// B fragments are 32B ds_loads. OOB A rows are clamped (not zeroed) -- they only feed
// OOB output rows, which the store guard drops.
__global__ void __launch_bounds__(256)
k_wmma_gemm(const _Float16* __restrict__ A, const _Float16* __restrict__ Bt,
            const float* __restrict__ bias, float* __restrict__ D_,
            int M, int K, int Ncol) {
    __shared__ _Float16 sB[64 * 256];          // 32KB max (K<=256)
    const int tn0 = blockIdx.y * 64;

    // ---- async copy Bt[tn0*K .. (tn0+64)*K) into LDS ----
    {
        const _Float16* src = Bt + (size_t)tn0 * K;
        const int chunks = (64 * K * 2) / 16;  // 16B chunks (1024 or 2048)
        unsigned lds_base = (unsigned)(uintptr_t)(&sB[0]);  // generic low 32 bits = LDS offset
        for (int c = threadIdx.x; c < chunks; c += 256) {
            unsigned loff = lds_base + (unsigned)c * 16u;
            unsigned long long gaddr = (unsigned long long)(uintptr_t)src + (size_t)c * 16u;
            asm volatile("global_load_async_to_lds_b128 %0, %1, off"
                         :: "v"(loff), "v"(gaddr) : "memory");
        }
        asm volatile("s_wait_asynccnt 0x0" ::: "memory");
    }
    __syncthreads();

    const int wid  = threadIdx.x >> 5;
    const int lane = threadIdx.x & 31;
    const int tm = (blockIdx.x * 8 + wid) * 16;
    if (tm >= M) return;

    const int l16   = lane & 15;     // A: row-in-tile, B/C: col-in-tile
    const int khalf = lane >> 4;     // 0/1
    int arow = tm + l16;
    if (arow >= M) arow = M - 1;     // clamp: garbage only feeds unstored rows
    const _Float16* arp = A + (size_t)arow * K;

    v8f acc[4];
    #pragma unroll
    for (int t = 0; t < 4; ++t) {
        float bv = bias[tn0 + t * 16 + l16];    // bias depends only on output column
        #pragma unroll
        for (int i = 0; i < 8; ++i) acc[t][i] = bv;
    }

    for (int k0 = 0; k0 < K; k0 += 32) {
        // A 16x32 f16 layout: lane holds 16 consecutive K in [k0+khalf*16, +16)
        v8h a0 = *(const v8h*)(arp + k0 + khalf * 16);
        v8h a1 = *(const v8h*)(arp + k0 + khalf * 16 + 8);
        v16h a;
        #pragma unroll
        for (int i = 0; i < 8; ++i) { a[i] = a0[i]; a[i + 8] = a1[i]; }
        #pragma unroll
        for (int t = 0; t < 4; ++t) {
            // B 32x16: lane holds 16 consecutive K of column (t*16+l16), from LDS
            v16h b = *(const v16h*)(&sB[(size_t)(t * 16 + l16) * K + k0 + khalf * 16]);
            acc[t] = __builtin_amdgcn_wmma_f32_16x16x32_f16(false, a, false, b,
                                                            (short)0, acc[t],
                                                            false, false);
        }
    }

    #pragma unroll
    for (int t = 0; t < 4; ++t) {
        #pragma unroll
        for (int i = 0; i < 8; ++i) {
            int row = tm + i + (khalf << 3);   // C/D: VGPR i -> M=i (lanes 0-15), M=i+8 (16-31)
            if (row < M) D_[(size_t)row * Ncol + tn0 + t * 16 + l16] = acc[t][i];
        }
    }
}

// ---------------- host ----------------
extern "C" void kernel_launch(void* const* d_in, const int* in_sizes, int n_in,
                              void* d_out, int out_size, void* d_ws, size_t ws_size,
                              hipStream_t stream) {
    const int*   x        = (const int*)d_in[0];
    const int*   eidx     = (const int*)d_in[1];
    const int*   eattr    = (const int*)d_in[2];
    const int*   batch    = (const int*)d_in[3];
    const float* atom_emb = (const float*)d_in[4];
    const float* bond_emb = (const float*)d_in[5];
    const float* vn_emb   = (const float*)d_in[6];
    const float* conv_W1  = (const float*)d_in[7];
    const float* conv_b1  = (const float*)d_in[8];
    const float* conv_g1  = (const float*)d_in[9];
    const float* conv_bb1 = (const float*)d_in[10];
    const float* conv_W2  = (const float*)d_in[11];
    const float* conv_b2  = (const float*)d_in[12];
    const float* norm_g   = (const float*)d_in[13];
    const float* norm_b   = (const float*)d_in[14];
    const float* vn_W1    = (const float*)d_in[15];
    const float* vn_b1    = (const float*)d_in[16];
    const float* vn_g     = (const float*)d_in[17];
    const float* vn_bb    = (const float*)d_in[18];
    const float* vn_W2    = (const float*)d_in[19];
    const float* vn_b2    = (const float*)d_in[20];
    float* out = (float*)d_out;

    char* wp = (char*)d_ws;
    auto alloc = [&](size_t bytes) {
        char* p = wp;
        wp += (bytes + 255) & ~(size_t)255;
        return p;
    };
    const size_t NC = (size_t)NN * CC;
    float*    h       = (float*)alloc(NC * 4);
    float*    h2      = (float*)alloc(NC * 4);
    float*    hnew    = (float*)alloc(NC * 4);
    float*    m       = (float*)alloc(NC * 4);
    float*    mx      = (float*)alloc(NC * 4);
    float*    den     = (float*)alloc(NC * 4);
    float*    z1      = (float*)alloc(NC * 2 * 4);       // [N, 2C] GEMM1 out
    _Float16* a16     = (_Float16*)alloc(NC * 2 * 2);    // f16 activations (<=256 wide)
    _Float16* a16b    = (_Float16*)alloc(NC * 2 * 2);
    _Float16* w16a    = (_Float16*)alloc((size_t)CC * 2 * CC * 2);
    _Float16* w16b    = (_Float16*)alloc((size_t)CC * 2 * CC * 2);
    float*    edgeemb = (float*)alloc((size_t)EE * CC * 4);
    float*    vn      = (float*)alloc((size_t)NGG * CC * 4);
    float*    vtmp    = (float*)alloc((size_t)NGG * CC * 4);
    float*    vz      = (float*)alloc((size_t)NGG * CC * 4);
    _Float16* v16     = (_Float16*)alloc((size_t)NGG * CC * 2);
    _Float16* v16b    = (_Float16*)alloc((size_t)NGG * CC * 2);

    const int* src = eidx;
    const int* dst = eidx + EE;

    const int TB = 256;
    const int gNC   = (int)((NC + TB - 1) / TB);
    const int gEC   = (int)(((size_t)EE * CC + TB - 1) / TB);
    const int EC4   = EE * (CC / 4);
    const int gEC4  = (EC4 + TB - 1) / TB;
    const int gNGC  = (NGG * CC + TB - 1) / TB;

    // --- encoders ---
    k_atom_encode<<<gNC, TB, 0, stream>>>(x, atom_emb, vn_emb, h);
    k_bond_encode<<<gEC, TB, 0, stream>>>(eattr, bond_emb, edgeemb);
    k_bcast_vn<<<gNGC, TB, 0, stream>>>(vn_emb, vn);

    auto gemm = [&](const _Float16* A, const _Float16* Bt, const float* bias,
                    float* D_, int M, int K, int Ncol) {
        dim3 grid((unsigned)(((M + 15) / 16 + 7) / 8), (unsigned)(Ncol / 64));
        k_wmma_gemm<<<grid, TB, 0, stream>>>(A, Bt, bias, D_, M, K, Ncol);
    };

    auto gen_conv = [&](int l, const float* hx, float* dest) {
        k_fill<<<gNC, TB, 0, stream>>>(mx, 0.f, (int)NC);
        k_fill<<<gNC, TB, 0, stream>>>(den, 0.f, (int)NC);
        k_fill<<<gNC, TB, 0, stream>>>(m, 0.f, (int)NC);
        k_msg_max<<<gEC4, TB, 0, stream>>>(hx, edgeemb, src, dst, (int*)mx, EC4);
        k_den<<<gEC4, TB, 0, stream>>>(hx, edgeemb, src, dst, mx, den, EC4);
        k_msum<<<gEC4, TB, 0, stream>>>(hx, edgeemb, src, dst, mx, den, m, EC4);
        // z = hx + m -> f16
        k_add_convert<<<gNC, TB, 0, stream>>>(hx, m, a16, (int)NC);
        // GEMM1: [N,128] x [128,256]
        k_wt<<<(CC * 2 * CC + TB - 1) / TB, TB, 0, stream>>>(
            conv_W1 + (size_t)l * CC * 2 * CC, w16a, CC, 2 * CC);
        gemm(a16, w16a, conv_b1 + (size_t)l * 2 * CC, z1, NN, CC, 2 * CC);
        // LN + ReLU over 256, straight to f16
        k_ln<<<(NN + 7) / 8, TB, 0, stream>>>(z1, conv_g1 + (size_t)l * 2 * CC,
                                              conv_bb1 + (size_t)l * 2 * CC,
                                              (float*)nullptr, a16b, NN, 2 * CC, 1);
        // GEMM2: [N,256] x [256,128]
        k_wt<<<(CC * 2 * CC + TB - 1) / TB, TB, 0, stream>>>(
            conv_W2 + (size_t)l * 2 * CC * CC, w16b, 2 * CC, CC);
        gemm(a16b, w16b, conv_b2 + (size_t)l * CC, dest, NN, 2 * CC, CC);
    };

    // layer 0
    gen_conv(0, h, hnew);
    hipMemcpyAsync(h, hnew, NC * 4, hipMemcpyDeviceToDevice, stream);

    for (int l = 1; l < LL; ++l) {
        // h2 = relu(LN(h))
        k_ln<<<(NN + 7) / 8, TB, 0, stream>>>(h, norm_g + (size_t)(l - 1) * CC,
                                              norm_b + (size_t)(l - 1) * CC,
                                              h2, (_Float16*)nullptr, NN, CC, 1);
        // vtmp = segsum(h2, batch) + vn
        hipMemcpyAsync(vtmp, vn, (size_t)NGG * CC * 4, hipMemcpyDeviceToDevice, stream);
        k_segsum<<<gNC, TB, 0, stream>>>(h2, batch, vtmp, (int)NC);
        // vn MLP (512x128 GEMMs through WMMA too)
        k_convert<<<gNGC, TB, 0, stream>>>(vtmp, v16, NGG * CC);
        k_wt<<<(CC * CC + TB - 1) / TB, TB, 0, stream>>>(
            vn_W1 + (size_t)(l - 1) * CC * CC, w16a, CC, CC);
        gemm(v16, w16a, vn_b1 + (size_t)(l - 1) * CC, vz, NGG, CC, CC);
        k_ln<<<(NGG + 7) / 8, TB, 0, stream>>>(vz, vn_g + (size_t)(l - 1) * CC,
                                               vn_bb + (size_t)(l - 1) * CC,
                                               (float*)nullptr, v16b, NGG, CC, 1);
        k_wt<<<(CC * CC + TB - 1) / TB, TB, 0, stream>>>(
            vn_W2 + (size_t)(l - 1) * CC * CC, w16b, CC, CC);
        gemm(v16b, w16b, vn_b2 + (size_t)(l - 1) * CC, vn, NGG, CC, CC);
        // h2 += vn[batch]
        k_add_gather<<<gNC, TB, 0, stream>>>(h2, vn, batch, (int)NC);
        // h = gen_conv(l, h2) + h
        gen_conv(l, h2, hnew);
        k_add_inplace<<<gNC, TB, 0, stream>>>(h, hnew, (int)NC);
    }

    // final LN (no relu) into h2, then global add pool
    k_ln<<<(NN + 7) / 8, TB, 0, stream>>>(h, norm_g + (size_t)(LL - 1) * CC,
                                          norm_b + (size_t)(LL - 1) * CC,
                                          h2, (_Float16*)nullptr, NN, CC, 0);
    k_fill<<<gNGC, TB, 0, stream>>>(out, 0.f, NGG * CC);
    k_segsum<<<gNC, TB, 0, stream>>>(h2, batch, out, (int)NC);
}
